// PointTransformerLayer_23313082483294
// MI455X (gfx1250) — matrix operand order repaired and checked by reading, not compiled, not run on hardware
//
#include <hip/hip_runtime.h>
#include <cmath>

// ---------------------------------------------------------------------------
// Problem constants (from reference): B=4, N=4096, C=OUT=HID=256, K=16
// ---------------------------------------------------------------------------
#define PT_B   4
#define PT_N   4096
#define PT_K   16
#define PT_M   (PT_B * PT_N)        // 16384 rows for all row-major GEMMs

typedef _Float16 v16h __attribute__((ext_vector_type(16)));
typedef _Float16 v8h  __attribute__((ext_vector_type(8)));
typedef _Float16 h2   __attribute__((ext_vector_type(2)));
typedef float    v8f  __attribute__((ext_vector_type(8)));
typedef unsigned int uint4v __attribute__((ext_vector_type(4)));
typedef int      int4v  __attribute__((ext_vector_type(4)));
typedef int      int8v  __attribute__((ext_vector_type(8)));

#define INF_F 3.0e38f

// --- CDNA5 TDM availability (device pass only; host pass parses fallback) ---
#if defined(__gfx1250__) && __has_builtin(__builtin_amdgcn_tensor_load_to_lds) \
                         && __has_builtin(__builtin_amdgcn_s_wait_tensorcnt)
#define USE_TDM 1
#else
#define USE_TDM 0
#endif
// toolchain discriminator: therock-10.0 headers ship the 6-arg TDM builtin
#if __has_include(<hip/amd_detail/amd_gfx1250_TDM.h>)
#define TDM_ARG6 1
#else
#define TDM_ARG6 0
#endif

// packed f32 pair -> f16 pair (v_cvt_pk_f16_f32)
__device__ __forceinline__ h2 pk2(float x, float y) {
    return __builtin_bit_cast(h2, __builtin_amdgcn_cvt_pkrtz(x, y));
}

#if USE_TDM
// low 32 bits of a flat pointer to LDS == LDS byte offset (ISA 10.2 aperture map)
__device__ __forceinline__ unsigned lds_off(const void* p) {
    return (unsigned)(unsigned long long)(size_t)p;
}
// 2D f16 tile load via Tensor Data Mover.  tile1 rows x tile0 halves,
// row stride stride0 halves; LDS padding encoded in g1w0.
__device__ __forceinline__ void tdm_load_2d(unsigned ldsOff,
                                            unsigned long long gaddr,
                                            unsigned g1w0,
                                            unsigned dim0, unsigned dim1,
                                            unsigned tile0, unsigned tile1,
                                            unsigned stride0)
{
    uint4v g0 = { 1u,                                       // count=1, user mode
                  ldsOff,
                  (unsigned)(gaddr & 0xffffffffull),
                  (unsigned)((gaddr >> 32) & 0x01ffffffull) | (2u << 30) }; // type=2
    int8v g1 = { (int)g1w0,
                 (int)(dim0 << 16),                          // tensor_dim0[15:0]
                 (int)((dim0 >> 16) | (dim1 << 16)),         // dim0 hi | dim1 lo
                 (int)((dim1 >> 16) | (tile0 << 16)),        // dim1 hi | tile_dim0
                 (int)(tile1 & 0xffff),                      // tile_dim1 (tile_dim2=0)
                 (int)stride0,                               // tensor_dim0_stride lo
                 0, 0 };
    int4v g2 = {};
    int4v g3 = {};
#if TDM_ARG6
    int8v gz = {};
    __builtin_amdgcn_tensor_load_to_lds(g0, g1, g2, g3, gz, 0);
#else
    __builtin_amdgcn_tensor_load_to_lds(g0, g1, g2, g3, 0);
#endif
}
// group1 word0: data_size=2B, pad_enable, pad_interval code, pad_amount code
#define TDM_W0(pintv, pamt) ((1u << 16) | (1u << 20) | ((pintv) << 22) | ((pamt) << 25))
#endif // USE_TDM

// ---------------------------------------------------------------------------
// Kernel 0: 256x256 f32 -> f16 transpose (weights, done once; L2-resident)
// dst[n][k] = (f16) src[k][n]
// ---------------------------------------------------------------------------
__global__ __launch_bounds__(256)
void transpose_to_f16(const float* __restrict__ src, _Float16* __restrict__ dst)
{
    __shared__ float t[16][17];
    const int tx = threadIdx.x & 15, ty = threadIdx.x >> 4;
    const int x0 = blockIdx.x * 16, y0 = blockIdx.y * 16;
    t[ty][tx] = src[(size_t)(y0 + ty) * 256 + x0 + tx];
    __syncthreads();
    dst[(size_t)(x0 + ty) * 256 + y0 + tx] = (_Float16)t[tx][ty];
}

// ---------------------------------------------------------------------------
// Kernel 1: kNN (K=16).  One wave32 per query; per-lane top-16 then a
// cooperative 16-pass argmin merge in LDS.
// ---------------------------------------------------------------------------
__global__ __launch_bounds__(256)
void knn_kernel(const float* __restrict__ pos,
                const unsigned char* __restrict__ mask,
                int* __restrict__ idxOut)
{
    __shared__ float mD[8][512];
    __shared__ int   mI[8][512];

    const int wave = threadIdx.x >> 5;
    const int lane = threadIdx.x & 31;
    const int q    = blockIdx.x * 8 + wave;
    const int b    = q >> 12;
    const int n    = q & (PT_N - 1);

    const float* pb = pos + (size_t)b * PT_N * 3;
    const float qx = pb[n * 3 + 0];
    const float qy = pb[n * 3 + 1];
    const float qz = pb[n * 3 + 2];
    const unsigned char* mrow = mask + ((size_t)b * PT_N + n) * PT_N;

    float bd[PT_K];
    int   bi[PT_K];
#pragma unroll
    for (int j = 0; j < PT_K; ++j) { bd[j] = INF_F; bi[j] = -1; }
    float worst = INF_F; int wpos = 0;

    for (int c = lane; c < PT_N; c += 32) {
        const float dx = qx - pb[c * 3 + 0];
        const float dy = qy - pb[c * 3 + 1];
        const float dz = qz - pb[c * 3 + 2];
        float d2 = dx * dx + dy * dy + dz * dz;
        if (!mrow[c]) d2 = INF_F;
        if (d2 < worst) {
            bd[wpos] = d2; bi[wpos] = c;
            worst = bd[0]; wpos = 0;
#pragma unroll
            for (int j = 1; j < PT_K; ++j)
                if (bd[j] > worst) { worst = bd[j]; wpos = j; }
        }
    }

#pragma unroll
    for (int j = 0; j < PT_K; ++j) {
        mD[wave][lane * PT_K + j] = bd[j];
        mI[wave][lane * PT_K + j] = bi[j];
    }
    __syncthreads();

    for (int pass = 0; pass < PT_K; ++pass) {
        float d = INF_F; int ci = 0x7fffffff; int p = -1;
#pragma unroll
        for (int s = 0; s < 16; ++s) {
            const int e = lane + s * 32;
            const float dd = mD[wave][e];
            const int   ii = mI[wave][e];
            if (dd < d || (dd == d && ii < ci)) { d = dd; ci = ii; p = e; }
        }
#pragma unroll
        for (int off = 16; off >= 1; off >>= 1) {
            const float od = __shfl_xor(d, off, 32);
            const int   oc = __shfl_xor(ci, off, 32);
            const int   op = __shfl_xor(p, off, 32);
            if (od < d || (od == d && oc < ci)) { d = od; ci = oc; p = op; }
        }
        if (lane == 0) {
            idxOut[(size_t)q * PT_K + pass] = (ci == 0x7fffffff) ? 0 : ci;
            if (p >= 0) mD[wave][p] = INF_F;
        }
        __syncthreads();
    }
}

// ---------------------------------------------------------------------------
// Kernel 2: out = act(A @ W + bias) (+residual), K=N=256, f16 WMMA, f32 acc.
// Wt: pre-transposed f16 weights, (N=256 rows) x (K=256 cols).
// One TDM tensor_load pulls the whole 256K x 64N panel (pad = +16 halves/row)
// into LDS once; the K loop then has NO barriers and NO staging.
// 128 threads = 4 waves; wave -> 16 rows x 64 cols (4 accs, shared A frag).
// ---------------------------------------------------------------------------
template <bool RELU>
__global__ __launch_bounds__(128)
void gemm256(const float* __restrict__ A, const _Float16* __restrict__ Wt,
             const float* __restrict__ bias, const float* __restrict__ resid,
             float* __restrict__ out, int M)
{
    __shared__ __align__(16) _Float16 Bsh[64][272];   // [n][k], 272 = 256+16 pad

    const int tid  = threadIdx.x;
    const int wave = tid >> 5;
    const int lane = tid & 31;
    const int half = lane >> 4;
    const int l16  = lane & 15;
    const int m0   = blockIdx.y * 64 + wave * 16;
    const int n0   = blockIdx.x * 64;

#if USE_TDM
    if (tid < 32) {
        // rows n0..n0+63 of Wt, each 256 halves; pad 8 DWORDs per 128 DWORDs
        tdm_load_2d(lds_off(&Bsh[0][0]),
                    (unsigned long long)(size_t)(Wt + (size_t)n0 * 256),
                    TDM_W0(6u, 7u), 256u, 256u, 256u, 64u, 256u);
        __builtin_amdgcn_s_wait_tensorcnt(0);
    }
#else
    for (int e = tid; e < 64 * 128; e += 128) {       // 64 rows x 128 u32
        const int r = e >> 7, c = e & 127;
        ((unsigned*)&Bsh[r][0])[c] =
            ((const unsigned*)(Wt + ((size_t)(n0 + r)) * 256))[c];
    }
#endif
    __syncthreads();

    v8f acc[4] = {};

    for (int kb = 0; kb < 256; kb += 32) {
        // A fragment: packed f32->f16 conversion, 4x float4 loads
        union { v16h v; h2 p[8]; } au;
        {
            const float* ar = A + (size_t)(m0 + l16) * 256 + kb + half * 8;
            const float4 f0 = *reinterpret_cast<const float4*>(ar);
            const float4 f1 = *reinterpret_cast<const float4*>(ar + 4);
            const float4 f2 = *reinterpret_cast<const float4*>(ar + 16);
            const float4 f3 = *reinterpret_cast<const float4*>(ar + 20);
            au.p[0] = pk2(f0.x, f0.y); au.p[1] = pk2(f0.z, f0.w);
            au.p[2] = pk2(f1.x, f1.y); au.p[3] = pk2(f1.z, f1.w);
            au.p[4] = pk2(f2.x, f2.y); au.p[5] = pk2(f2.z, f2.w);
            au.p[6] = pk2(f3.x, f3.y); au.p[7] = pk2(f3.z, f3.w);
        }

#pragma unroll
        for (int t = 0; t < 4; ++t) {
            union { v16h v; v8h h[2]; } bu;
            const _Float16* br = &Bsh[t * 16 + l16][kb + half * 8];
            bu.h[0] = *reinterpret_cast<const v8h*>(br);       // K = kb+half*8+0..7
            bu.h[1] = *reinterpret_cast<const v8h*>(br + 16);  // K = kb+16+half*8+0..7
            acc[t] = __builtin_amdgcn_wmma_f32_16x16x32_f16(
                false, au.v, false, bu.v, (short)0, acc[t], false, false);
        }
    }

#pragma unroll
    for (int t = 0; t < 4; ++t) {
#pragma unroll
        for (int r = 0; r < 8; ++r) {
            const int row = m0 + half * 8 + r;
            const int col = n0 + t * 16 + l16;
            float v = acc[t][r] + bias[col];
            if (RELU) v = fmaxf(v, 0.0f);
            if (resid) v += resid[(size_t)row * 256 + col];
            out[(size_t)row * 256 + col] = v;
        }
    }
}

// ---------------------------------------------------------------------------
// Kernel 3: fused polar pos-encoding + neighborhood attention.
// One 256-thread block per point.  pe layer2 = (16x256)@(256x256) WMMA with
// pW2t K-panels (256 rows x 32 halves) double-buffered via TDM.
// ---------------------------------------------------------------------------
__global__ __launch_bounds__(256)
void attn_pe_kernel(const float* __restrict__ pos,
                    const float* __restrict__ normal,
                    const int* __restrict__ idx,
                    const float* __restrict__ qbuf,
                    const float* __restrict__ kbuf,
                    const float* __restrict__ vbuf,
                    const float* __restrict__ pW1, const float* __restrict__ pb1,
                    const _Float16* __restrict__ pW2t,
                    const float* __restrict__ pb2,
                    float* __restrict__ outbuf)
{
    __shared__ float    kgS[PT_K][256];
    __shared__ float    vgS[PT_K][256];
    __shared__ float    qS[256];
    __shared__ __align__(16) _Float16 hidH[PT_K][264];   // [m][k], padded
    __shared__ __align__(16) _Float16 BshP[2][256][40];  // [buf][n][k], 40=32+8 pad
    __shared__ float    rL[PT_K], thL[PT_K];
    __shared__ int      idxL[PT_K];
    __shared__ float    scoreL[PT_K];

    const int pt  = blockIdx.x;
    const int b   = pt >> 12;
    const int n   = pt & (PT_N - 1);
    const int tid = threadIdx.x;

    // --- neighbor geometry: r, theta (threads 0..15) ---
    if (tid < PT_K) {
        const int nb = idx[(size_t)pt * PT_K + tid];
        idxL[tid] = nb;
        const size_t bn  = (size_t)b * PT_N + n;
        const size_t bnb = (size_t)b * PT_N + nb;
        float nx = normal[bn * 3 + 0], ny = normal[bn * 3 + 1], nz = normal[bn * 3 + 2];
        const float nn = fmaxf(sqrtf(nx * nx + ny * ny + nz * nz), 1e-12f);
        nx /= nn; ny /= nn; nz /= nn;
        const float px = pos[bn * 3 + 0] - pos[bnb * 3 + 0];
        const float py = pos[bn * 3 + 1] - pos[bnb * 3 + 1];
        const float pz = pos[bn * 3 + 2] - pos[bnb * 3 + 2];
        const float sqr = px * px + py * py + pz * pz;
        const float r   = (sqr > 0.0f) ? sqrtf(sqr) : 0.0f;
        const float dot = px * nx + py * ny + pz * nz;
        float arg = dot / (r + 1e-8f);
        arg = fminf(1.0f, fmaxf(-1.0f, arg));
        rL[tid]  = r;
        thL[tid] = acosf(arg);
    }
#if USE_TDM
    if (tid < 32) {   // prime slab 0 while geometry settles
        tdm_load_2d(lds_off(&BshP[0][0][0]),
                    (unsigned long long)(size_t)pW2t,
                    TDM_W0(3u, 3u), 256u, 256u, 32u, 256u, 256u);
    }
#endif
    __syncthreads();

    // --- gather k/v rows of the 16 neighbors + q row ---
    {
        const int m  = tid >> 4;
        const int cb = (tid & 15) * 16;
        const size_t rowOff = ((size_t)b * PT_N + idxL[m]) * 256 + cb;
        const float4* ks = reinterpret_cast<const float4*>(kbuf + rowOff);
        const float4* vs = reinterpret_cast<const float4*>(vbuf + rowOff);
#pragma unroll
        for (int j = 0; j < 4; ++j) {
            const float4 kv = ks[j];
            kgS[m][cb + j * 4 + 0] = kv.x; kgS[m][cb + j * 4 + 1] = kv.y;
            kgS[m][cb + j * 4 + 2] = kv.z; kgS[m][cb + j * 4 + 3] = kv.w;
            const float4 vv = vs[j];
            vgS[m][cb + j * 4 + 0] = vv.x; vgS[m][cb + j * 4 + 1] = vv.y;
            vgS[m][cb + j * 4 + 2] = vv.z; vgS[m][cb + j * 4 + 3] = vv.w;
        }
        if (tid < 64) {
            const float4 qv = reinterpret_cast<const float4*>(
                qbuf + (size_t)pt * 256)[tid];
            qS[tid * 4 + 0] = qv.x; qS[tid * 4 + 1] = qv.y;
            qS[tid * 4 + 2] = qv.z; qS[tid * 4 + 3] = qv.w;
        }
    }

    // --- pe layer 1 (K-dim is 2 -> VALU): hid[m][h], f16, [m][k] layout ---
    {
        const int h  = tid;
        const float w0 = pW1[h];
        const float w1 = pW1[256 + h];
        const float bb = pb1[h];
#pragma unroll
        for (int m = 0; m < PT_K; ++m) {
            const float hv = fmaxf(rL[m] * w0 + thL[m] * w1 + bb, 0.0f);
            hidH[m][h] = (_Float16)hv;
        }
    }

    // --- pe layer 2: WMMA over 8 K-slabs, double-buffered pW2t panels ---
    const int wave  = tid >> 5;
    const int lane  = tid & 31;
    const int halfq = lane >> 4;
    const int l16   = lane & 15;
    v8f acc[2] = {};

    for (int s = 0; s < 8; ++s) {
        const int kb = s * 32;
#if USE_TDM
        if (tid < 32) {
            if (s + 1 < 8) {
                tdm_load_2d(lds_off(&BshP[(s + 1) & 1][0][0]),
                            (unsigned long long)(size_t)(pW2t + (size_t)(s + 1) * 32),
                            TDM_W0(3u, 3u), 256u, 256u, 32u, 256u, 256u);
                __builtin_amdgcn_s_wait_tensorcnt(1);   // slab s complete
            } else {
                __builtin_amdgcn_s_wait_tensorcnt(0);
            }
        }
#else
        for (int e = tid; e < 256 * 16; e += 256) {     // 256 rows x 16 u32
            const int r = e >> 4, c = e & 15;
            ((unsigned*)&BshP[s & 1][r][0])[c] =
                ((const unsigned*)(pW2t + (size_t)r * 256 + kb))[c];
        }
#endif
        __syncthreads();

        union { v16h v; v8h h[2]; } au;
        {
            const _Float16* arow = &hidH[l16][kb + halfq * 8];
            au.h[0] = *reinterpret_cast<const v8h*>(arow);        // K=kb+half*8+0..7
            au.h[1] = *reinterpret_cast<const v8h*>(arow + 16);   // K=kb+16+half*8..
        }
#pragma unroll
        for (int tt = 0; tt < 2; ++tt) {
            const int tile = wave * 2 + tt;
            union { v16h v; v8h h[2]; } bu;
            const _Float16* br = &BshP[s & 1][tile * 16 + l16][halfq * 8];
            bu.h[0] = *reinterpret_cast<const v8h*>(br);
            bu.h[1] = *reinterpret_cast<const v8h*>(br + 16);
            acc[tt] = __builtin_amdgcn_wmma_f32_16x16x32_f16(
                false, au.v, false, bu.v, (short)0, acc[tt], false, false);
        }
        __syncthreads();
    }

    // --- pe += bias ; kg += pe ; vg += pe ---
#pragma unroll
    for (int tt = 0; tt < 2; ++tt) {
#pragma unroll
        for (int r = 0; r < 8; ++r) {
            const int mrow = halfq * 8 + r;
            const int col  = (wave * 2 + tt) * 16 + l16;
            const float pe = acc[tt][r] + pb2[col];
            kgS[mrow][col] += pe;
            vgS[mrow][col] += pe;
        }
    }
    __syncthreads();

    // --- scores_j = (q . kg_j) / 16 ---
    {
        const int j  = tid >> 4;
        const int cb = (tid & 15) * 16;
        float s = 0.0f;
#pragma unroll
        for (int c = 0; c < 16; ++c)
            s += qS[cb + c] * kgS[j][cb + c];
#pragma unroll
        for (int off = 8; off >= 1; off >>= 1)
            s += __shfl_xor(s, off, 32);
        if ((tid & 15) == 0) scoreL[j] = s * 0.0625f;
    }
    __syncthreads();

    // --- softmax over 16 + weighted sum of vg ---
    {
        float mx = -INF_F;
#pragma unroll
        for (int j = 0; j < PT_K; ++j) mx = fmaxf(mx, scoreL[j]);
        float e[PT_K], sum = 0.0f;
#pragma unroll
        for (int j = 0; j < PT_K; ++j) { e[j] = __expf(scoreL[j] - mx); sum += e[j]; }
        const float inv = 1.0f / sum;
        float o = 0.0f;
#pragma unroll
        for (int j = 0; j < PT_K; ++j) o += e[j] * vgS[j][tid];
        outbuf[(size_t)pt * 256 + tid] = o * inv;
    }
}

// ---------------------------------------------------------------------------
// Kernel 4: LayerNorm over last dim (256). One wave per row.
// ---------------------------------------------------------------------------
__global__ __launch_bounds__(256)
void layernorm_kernel(const float* __restrict__ in,
                      const float* __restrict__ g,
                      const float* __restrict__ bta,
                      float* __restrict__ out)
{
    const int wave = threadIdx.x >> 5;
    const int lane = threadIdx.x & 31;
    const size_t row = (size_t)blockIdx.x * 8 + wave;
    const float* x = in + row * 256;

    float xv[8], s = 0.0f, sq = 0.0f;
#pragma unroll
    for (int j = 0; j < 8; ++j) {
        xv[j] = x[lane + 32 * j];
        s  += xv[j];
        sq += xv[j] * xv[j];
    }
#pragma unroll
    for (int off = 16; off >= 1; off >>= 1) {
        s  += __shfl_xor(s, off, 32);
        sq += __shfl_xor(sq, off, 32);
    }
    const float mu  = s * (1.0f / 256.0f);
    const float var = sq * (1.0f / 256.0f) - mu * mu;
    const float rs  = rsqrtf(var + 1e-6f);
#pragma unroll
    for (int j = 0; j < 8; ++j) {
        const int c = lane + 32 * j;
        out[row * 256 + c] = (xv[j] - mu) * rs * g[c] + bta[c];
    }
}

// ---------------------------------------------------------------------------
// Launch
// ---------------------------------------------------------------------------
extern "C" void kernel_launch(void* const* d_in, const int* in_sizes, int n_in,
                              void* d_out, int out_size, void* d_ws, size_t ws_size,
                              hipStream_t stream)
{
    const float* x      = (const float*)d_in[0];
    const float* pos    = (const float*)d_in[1];
    const float* normal = (const float*)d_in[2];
    const unsigned char* mask = (const unsigned char*)d_in[3];
    const float* qW1 = (const float*)d_in[4],  *qb1 = (const float*)d_in[5];
    const float* qW2 = (const float*)d_in[6],  *qb2 = (const float*)d_in[7];
    const float* kW1 = (const float*)d_in[8],  *kb1 = (const float*)d_in[9];
    const float* kW2 = (const float*)d_in[10], *kb2 = (const float*)d_in[11];
    const float* vW1 = (const float*)d_in[12], *vb1 = (const float*)d_in[13];
    const float* vW2 = (const float*)d_in[14], *vb2 = (const float*)d_in[15];
    const float* pW1 = (const float*)d_in[16], *pb1 = (const float*)d_in[17];
    const float* pW2 = (const float*)d_in[18], *pb2 = (const float*)d_in[19];
    const float* oW1 = (const float*)d_in[20], *ob1 = (const float*)d_in[21];
    const float* oW2 = (const float*)d_in[22], *ob2 = (const float*)d_in[23];
    const float* ln_g = (const float*)d_in[24], *ln_b = (const float*)d_in[25];
    float* outp = (float*)d_out;

    // workspace layout
    char* ws = (char*)d_ws;
    const size_t SZ = (size_t)PT_M * 256 * sizeof(float);   // 16 MB per tensor
    int*   idxW  = (int*)ws;                                // 1 MB
    float* qbuf  = (float*)(ws + (1u << 20));
    float* kbuf  = (float*)(ws + (1u << 20) + 1 * SZ);
    float* vbuf  = (float*)(ws + (1u << 20) + 2 * SZ);
    float* hid   = (float*)(ws + (1u << 20) + 3 * SZ);
    float* attnO = (float*)(ws + (1u << 20) + 4 * SZ);
    float* fco   = (float*)(ws + (1u << 20) + 5 * SZ);
    _Float16* wt = (_Float16*)(ws + (1u << 20) + 6 * SZ);   // 9 x 128 KB f16

    const float* wsrc[9] = { qW1, qW2, kW1, kW2, vW1, vW2, oW1, oW2, pW2 };
    for (int i = 0; i < 9; ++i)
        transpose_to_f16<<<dim3(16, 16), 256, 0, stream>>>(
            wsrc[i], wt + (size_t)i * 256 * 256);
    _Float16* qW1t = wt + 0 * 65536, *qW2t = wt + 1 * 65536;
    _Float16* kW1t = wt + 2 * 65536, *kW2t = wt + 3 * 65536;
    _Float16* vW1t = wt + 4 * 65536, *vW2t = wt + 5 * 65536;
    _Float16* oW1t = wt + 6 * 65536, *oW2t = wt + 7 * 65536;
    _Float16* pW2t = wt + 8 * 65536;

    const dim3 gGemm(4, PT_M / 64);
    const int  M = PT_M;

    knn_kernel<<<PT_M / 8, 256, 0, stream>>>(pos, mask, idxW);

    gemm256<true ><<<gGemm, 128, 0, stream>>>(x,   qW1t, qb1, nullptr, hid,  M);
    gemm256<false><<<gGemm, 128, 0, stream>>>(hid, qW2t, qb2, nullptr, qbuf, M);
    gemm256<true ><<<gGemm, 128, 0, stream>>>(x,   kW1t, kb1, nullptr, hid,  M);
    gemm256<false><<<gGemm, 128, 0, stream>>>(hid, kW2t, kb2, nullptr, kbuf, M);
    gemm256<true ><<<gGemm, 128, 0, stream>>>(x,   vW1t, vb1, nullptr, hid,  M);
    gemm256<false><<<gGemm, 128, 0, stream>>>(hid, vW2t, vb2, nullptr, vbuf, M);

    attn_pe_kernel<<<PT_M, 256, 0, stream>>>(pos, normal, idxW,
                                             qbuf, kbuf, vbuf,
                                             pW1, pb1, pW2t, pb2, attnO);

    gemm256<true ><<<gGemm, 128, 0, stream>>>(attnO, oW1t, ob1, nullptr, hid, M);
    gemm256<false><<<gGemm, 128, 0, stream>>>(hid,   oW2t, ob2, qbuf,    fco, M);

    layernorm_kernel<<<PT_M / 8, 256, 0, stream>>>(fco, ln_g, ln_b, outp);
}
